// LinearLayer_76020921140428
// MI455X (gfx1250) — matrix operand order repaired and checked by reading, not compiled, not run on hardware
//
#include <hip/hip_runtime.h>

// CDNA5 / gfx1250 wave32 kernels for:
//   w_eff[o,i] = (sum_p coef[o,i,p]) * weights[o,i]   (phase 1, bandwidth bound)
//   out[b,o]   = sum_i x[b,i] * w_eff[o,i]            (phase 2, f16 WMMA GEMM)
//
// Phase 2: double-buffered LDS, ONE barrier per K-step (BK=64, 16 steps):
//   barrier -> issue async B (global->LDS, ASYNCcnt) -> 8 WMMAs (K chunk 0)
//   -> cvt+ds_store next A tile (x loads had a full chunk to land)
//   -> 8 WMMAs (K chunk 1) -> issue next A global loads -> s_wait_asynccnt
//
// d_ws usage: OUT*IN * sizeof(_Float16) = 2 MiB for the f16 effective weights.

typedef __attribute__((ext_vector_type(16))) _Float16 v16h;
typedef __attribute__((ext_vector_type(8)))  _Float16 v8h;
typedef __attribute__((ext_vector_type(8)))  float    v8f;

#define BATCH 8192
#define IN    1024
#define OUT   1024

#define BM 128
#define BN 128
#define BK 64
#define KT (IN / BK)   // 16 K-steps
// padded LDS row stride in halves: 144 bytes = 36 DWORDs -> 16 consecutive rows
// hit 16 distinct banks, and every 8-half chunk stays 16B aligned.
#define LDA 72
#define LDB 72

union V16 { v16h v; v8h h[2]; };

// ---------------- Phase 1: reduce P and fold weights, f32 -> f16 ----------------
__global__ __launch_bounds__(256)
void weff_f16_kernel(const float* __restrict__ coef,
                     const float* __restrict__ w,
                     _Float16* __restrict__ weff) {
  const size_t t = (size_t)blockIdx.x * blockDim.x + threadIdx.x; // one (o,i)
  const float4* c4 = reinterpret_cast<const float4*>(coef) + t * 2;
  float4 a = c4[0];
  float4 b = c4[1];
  float s = ((a.x + a.y) + (a.z + a.w)) + ((b.x + b.y) + (b.z + b.w));
  weff[t] = (_Float16)(s * w[t]);
}

// ---- helpers -------------------------------------------------------------------
__device__ inline void load_a_regs(const float* __restrict__ x, size_t base,
                                   float4 (&f)[8]) {
  const float4* xa = reinterpret_cast<const float4*>(&x[base]);
#pragma unroll
  for (int c = 0; c < 8; ++c) f[c] = xa[c];
}

__device__ inline void store_a_lds(_Float16* dst, const float4 (&f)[8]) {
#pragma unroll
  for (int c = 0; c < 4; ++c) {
    v8h p;
    p[0] = (_Float16)f[2 * c].x;     p[1] = (_Float16)f[2 * c].y;
    p[2] = (_Float16)f[2 * c].z;     p[3] = (_Float16)f[2 * c].w;
    p[4] = (_Float16)f[2 * c + 1].x; p[5] = (_Float16)f[2 * c + 1].y;
    p[6] = (_Float16)f[2 * c + 1].z; p[7] = (_Float16)f[2 * c + 1].w;
    *(v8h*)(dst + 8 * c) = p;
  }
}

// CDNA5 async copy: 64 bytes (4 x b128) global -> LDS, tracked by ASYNCcnt.
// LDS flat pointers: low 32 bits of the shared-aperture address are the LDS
// byte address (aperture tag lives in the high bits).
__device__ inline void async_b_lds(const _Float16* __restrict__ gsrc,
                                   _Float16* ldst) {
  unsigned l = (unsigned)(uintptr_t)ldst;
  unsigned long long g = (unsigned long long)(uintptr_t)gsrc;
#pragma unroll
  for (int c = 0; c < 4; ++c) {
    asm volatile("global_load_async_to_lds_b128 %0, %1, off"
                 :
                 : "v"(l + 16u * c), "v"(g + 16ull * c)
                 : "memory");
  }
}

// 16x16x32 fragment loads + 8 WMMAs for one 32-wide K chunk of the 64-wide tile.
__device__ inline void mma_chunk(const _Float16* __restrict__ la,
                                 const _Float16* __restrict__ lb,
                                 int wm, int wn, int r, int hs, int kbase,
                                 v8f (&acc)[2][4]) {
  V16 afrag[2], bfrag[4];
  // A fragment (16x32 f16): lane r = row; hs picks K 0-7/16-23 vs 8-15/24-31
#pragma unroll
  for (int tm = 0; tm < 2; ++tm) {
    const _Float16* ap = &la[(wm * 32 + tm * 16 + r) * LDA + kbase + hs * 8];
    afrag[tm].h[0] = *(const v8h*)(ap);
    afrag[tm].h[1] = *(const v8h*)(ap + 16);
  }
  // B fragment (32x16 f16): lane r = column N; hs picks K 0-7/16-23 vs 8-15/24-31
  // B[k][n] = weff[n_global][k_global] -> contiguous K run in the lds_b row
#pragma unroll
  for (int tn = 0; tn < 4; ++tn) {
    const _Float16* bp = &lb[(wn * 64 + tn * 16 + r) * LDB + kbase + hs * 8];
    bfrag[tn].h[0] = *(const v8h*)(bp);
    bfrag[tn].h[1] = *(const v8h*)(bp + 16);
  }
#pragma unroll
  for (int tm = 0; tm < 2; ++tm)
#pragma unroll
    for (int tn = 0; tn < 4; ++tn)
      acc[tm][tn] = __builtin_amdgcn_wmma_f32_16x16x32_f16(
          false, afrag[tm].v, false, bfrag[tn].v,
          (short)0, acc[tm][tn], false, false);
}

// ---------------- Phase 2: x[8192,1024] @ weff[1024,1024]^T -> out --------------
__global__ __launch_bounds__(256)
void gemm_wmma_f16_kernel(const float* __restrict__ x,
                          const _Float16* __restrict__ weff,
                          float* __restrict__ out) {
  __shared__ __align__(16) _Float16 lds_a[2][BM * LDA]; // x tile    [row][k]
  __shared__ __align__(16) _Float16 lds_b[2][BN * LDB]; // weff tile [o]  [k]

  const int tid  = threadIdx.x;
  const int lane = tid & 31;
  const int wave = tid >> 5;
  const int bm   = blockIdx.y;   // M tile (64)
  const int bn   = blockIdx.x;   // N tile (8)

  const int wm = wave & 3;       // 4 waves along M: 32 rows each
  const int wn = wave >> 2;      // 2 waves along N: 64 cols each

  // staging coordinates: 256 threads cover 128 rows x 64 cols, 32 elems/thread
  const int srow = tid >> 1;           // 0..127
  const int sseg = (tid & 1) * 32;     // 0 or 32 (elements)
  const size_t xbase = (size_t)(bm * BM + srow) * IN + sseg;
  const size_t wbase = (size_t)(bn * BN + srow) * IN + sseg;
  _Float16* const a_st0 = &lds_a[0][srow * LDA + sseg];
  _Float16* const a_st1 = &lds_a[1][srow * LDA + sseg];
  _Float16* const b_st0 = &lds_b[0][srow * LDB + sseg];
  _Float16* const b_st1 = &lds_b[1][srow * LDB + sseg];

  const v8f zero = {};
  v8f acc[2][4];
#pragma unroll
  for (int i = 0; i < 2; ++i)
#pragma unroll
    for (int j = 0; j < 4; ++j) acc[i][j] = zero;

  const int r  = lane & 15;   // row (A) / col (B,C/D) within 16x16 tile
  const int hs = lane >> 4;   // half-select

  // ---- prologue: stage K-step 0 into buffer 0 ----
  float4 areg[8];
  async_b_lds(&weff[wbase], b_st0);
  load_a_regs(x, xbase, areg);
  store_a_lds(a_st0, areg);

  for (int kt = 0; kt < KT; ++kt) {
    const int buf = kt & 1;
    const bool more = (kt + 1 < KT);

    if (more) {
      load_a_regs(x, xbase + (size_t)(kt + 1) * BK, areg);   // next A -> regs
      if (kt + 2 < KT)                                       // L2 warm-up
        __builtin_prefetch(&x[xbase + (size_t)(kt + 2) * BK], 0, 3);
    }

    // drain this wave's async B writes into the buffer being published,
    // then one barrier per K-step publishes `buf` and retires `buf^1` reads.
    asm volatile("s_wait_asynccnt 0x0" ::: "memory");
    __syncthreads();

    const _Float16* la = lds_a[buf];
    const _Float16* lb = lds_b[buf];

    // issue next B tile immediately (no register dependency): full step to land
    if (more) async_b_lds(&weff[wbase + (size_t)(kt + 1) * BK], buf ? b_st0 : b_st1);

    mma_chunk(la, lb, wm, wn, r, hs, 0, acc);        // K chunk 0 (8 WMMAs)

    // consume next A regs mid-step: x loads had a chunk of WMMAs to complete,
    // and the ds_stores get chunk 1 to drain before the next barrier.
    if (more) store_a_lds(buf ? a_st0 : a_st1, areg);

    mma_chunk(la, lb, wm, wn, r, hs, 32, acc);       // K chunk 1 (8 WMMAs)
  }

  // C/D layout: VGPR rr -> row rr + 8*hs, col = lane&15
#pragma unroll
  for (int tm = 0; tm < 2; ++tm) {
#pragma unroll
    for (int tn = 0; tn < 4; ++tn) {
      const int row0 = bm * BM + wm * 32 + tm * 16 + hs * 8;
      const int col  = bn * BN + wn * 64 + tn * 16 + r;
#pragma unroll
      for (int rr = 0; rr < 8; ++rr)
        out[(size_t)(row0 + rr) * OUT + col] = acc[tm][tn][rr];
    }
  }
}

extern "C" void kernel_launch(void* const* d_in, const int* in_sizes, int n_in,
                              void* d_out, int out_size, void* d_ws, size_t ws_size,
                              hipStream_t stream) {
  const float* x    = (const float*)d_in[0];  // [8192,1024]
  const float* coef = (const float*)d_in[1];  // [1024,1024,8]
  const float* w    = (const float*)d_in[2];  // [1024,1024]
  float* out        = (float*)d_out;          // [8192,1024]
  _Float16* weff    = (_Float16*)d_ws;        // [1024,1024] f16 (2 MiB)

  weff_f16_kernel<<<(OUT * IN) / 256, 256, 0, stream>>>(coef, w, weff);

  dim3 grid(OUT / BN, BATCH / BM);            // (8, 64)
  gemm_wmma_f16_kernel<<<grid, 256, 0, stream>>>(x, weff, out);
}
